// TransformerBlock_81939386073750
// MI455X (gfx1250) — matrix operand order repaired and checked by reading, not compiled, not run on hardware
//
#include <hip/hip_runtime.h>
#include <hip/hip_bf16.h>

// ---------------------------------------------------------------------------
// Types / helpers
// ---------------------------------------------------------------------------
typedef __attribute__((ext_vector_type(16))) __bf16 v16bf;
typedef __attribute__((ext_vector_type(8)))  float  v8f;
typedef __attribute__((ext_vector_type(4)))  unsigned int v4u;
typedef __attribute__((ext_vector_type(4)))  int v4i;
typedef __attribute__((ext_vector_type(8)))  int v8i;

struct alignas(16) U128 { unsigned int x, y, z, w; };
struct U256 { U128 lo, hi; };
union  Pack { U128 q[2]; unsigned short s[16]; };

// Tensor Data Mover availability (probe-confirmed builtins; arity differs by
// toolchain: ROCm7.2/clang-22 = 5 args, amdgpu-toolchain/clang-23 = 6 args).
#if defined(__has_builtin)
#if __has_builtin(__builtin_amdgcn_tensor_load_to_lds) && \
    __has_builtin(__builtin_amdgcn_s_wait_tensorcnt)
#define USE_TDM 1
#endif
#endif
#ifndef USE_TDM
#define USE_TDM 0
#endif

static __device__ __forceinline__ unsigned short f2bf(float f) {
    unsigned int u = __builtin_bit_cast(unsigned int, f);
    u += 0x7FFFu + ((u >> 16) & 1u);           // round-to-nearest-even
    return (unsigned short)(u >> 16);
}
static __device__ __forceinline__ float bf2f(unsigned short h) {
    unsigned int u = ((unsigned int)h) << 16;
    return __builtin_bit_cast(float, u);
}

// A-matrix 16x32 bf16 fragment from row-major LDS tile.
// ISA layout: lane<16 -> K[0..7] in v0-3, K[16..23] in v4-7 ; lane>=16 -> K[8..15], K[24..31]
static __device__ __forceinline__ v16bf frag_a(const unsigned short* lds, int row,
                                               int stride, int k0) {
    const int lane = threadIdx.x & 31;
    const int kOff = (lane < 16) ? 0 : 8;
    U256 t;
    t.lo = *(const U128*)(lds + row * stride + k0 + kOff);
    t.hi = *(const U128*)(lds + row * stride + k0 + kOff + 16);
    return __builtin_bit_cast(v16bf, t);
}

// B-matrix 32x16 bf16 fragment from K-contiguous (transposed) LDS tile:
// lds[col*stride + k] holds B[k][col].  lane<16 -> K[0..7],K[16..23]; lane>=16 -> K[8..15],K[24..31]
static __device__ __forceinline__ v16bf frag_b(const unsigned short* lds, int col,
                                               int stride, int k0) {
    const int lane = threadIdx.x & 31;
    const int kOff = (lane < 16) ? 0 : 8;
    U256 t;
    t.lo = *(const U128*)(lds + col * stride + k0 + kOff);
    t.hi = *(const U128*)(lds + col * stride + k0 + kOff + 16);
    return __builtin_bit_cast(v16bf, t);
}

static __device__ __forceinline__ v8f wmma_bf16(v16bf a, v16bf b, v8f c) {
    return __builtin_amdgcn_wmma_f32_16x16x32_bf16(false, a, false, b,
                                                   (short)0, c, false, false);
}

#if USE_TDM
// Issue a 2-D TDM tile load: copies tile_d1 rows of tile_d0 bf16 elements
// (global row stride = stride0 elements) into contiguous LDS rows.
// D# packing per CDNA5 ISA sec 8.3/8.4 (data_size=1 => 2-byte elements).
static __device__ __forceinline__ void tdm_load_2d(
    unsigned lds_byte_addr, const void* gaddr,
    unsigned tensor_d0, unsigned tensor_d1, unsigned long long stride0,
    unsigned tile_d0, unsigned tile_d1)
{
    unsigned long long ga = (unsigned long long)gaddr;
    v4u g0;
    g0.x = 1u;                                        // count=1, user mode
    g0.y = lds_byte_addr;                             // lds_addr (bytes)
    g0.z = (unsigned)(ga & 0xFFFFFFFFu);              // global_addr[31:0]
    g0.w = (unsigned)((ga >> 32) & 0x01FFFFFFu) | (2u << 30);  // addr[56:32] | type=2
    v8i g1;
    g1[0] = (int)(1u << 16);                          // wg_mask=0, data_size=1 (2B)
    g1[1] = (int)((tensor_d0 & 0xFFFFu) << 16);       // tensor_dim0[15:0] @ bits63:48
    g1[2] = (int)(((tensor_d0 >> 16) & 0xFFFFu) |     // tensor_dim0[31:16]
                  ((tensor_d1 & 0xFFFFu) << 16));     // tensor_dim1[15:0]
    g1[3] = (int)(((tensor_d1 >> 16) & 0xFFFFu) |     // tensor_dim1[31:16]
                  ((tile_d0 & 0xFFFFu) << 16));       // tile_dim0
    g1[4] = (int)(tile_d1 & 0xFFFFu);                 // tile_dim1 (tile_dim2=0)
    g1[5] = (int)(unsigned)(stride0 & 0xFFFFFFFFu);   // dim0_stride[31:0]
    g1[6] = (int)(unsigned)((stride0 >> 32) & 0xFFFFu); // dim0_stride[47:32]
    g1[7] = 0;
    v4i z4 = {0, 0, 0, 0};
#if __clang_major__ >= 23
    v8i z8 = {0, 0, 0, 0, 0, 0, 0, 0};
    __builtin_amdgcn_tensor_load_to_lds(g0, g1, z4, z4, z8, 0);
#else
    __builtin_amdgcn_tensor_load_to_lds(g0, g1, z4, z4, 0);
#endif
}
#endif

// ---------------------------------------------------------------------------
// Generic bf16 GEMM: C[M,N] = A[M,K] * B[K,N] + bias (+ resid)
// EPI==0: store bf16.  EPI==1: store f32 with residual add.
// Block tile 128x128, K-step 64, 8 waves (2x4), wave tile 64x32.
// A tile staged by the Tensor Data Mover (wave 0 issues, tensorcnt wait);
// B tile staged transposed by all threads.
// ---------------------------------------------------------------------------
template <int EPI>
__global__ __launch_bounds__(256) void gemm_bf16_kernel(
    const unsigned short* __restrict__ A, const unsigned short* __restrict__ B,
    const float* __restrict__ bias, const float* __restrict__ resid,
    void* __restrict__ Cout, int M, int N, int K)
{
    __shared__ unsigned short As[128 * 64];   // row-major [m][k]    16KB
    __shared__ unsigned short Bs[128 * 64];   // transposed [n][k]   16KB

    const int tid  = threadIdx.x;
    const int lane = tid & 31;
    const int wave = tid >> 5;
    const int wm   = wave >> 2;               // 0..1
    const int wn   = wave & 3;                // 0..3
    const int bm   = blockIdx.y * 128;
    const int bn   = blockIdx.x * 128;

    const v8f vz = {0.f, 0.f, 0.f, 0.f, 0.f, 0.f, 0.f, 0.f};
    v8f acc[4][2];
#pragma unroll
    for (int i = 0; i < 4; ++i)
#pragma unroll
        for (int j = 0; j < 2; ++j) acc[i][j] = vz;

    for (int k0 = 0; k0 < K; k0 += 64) {
        // ---- stage A tile 128x64 ----
#if USE_TDM
        if (tid < 32) {
            tdm_load_2d((unsigned)(unsigned long long)(void*)As,
                        A + (size_t)bm * K + k0,
                        64u, 128u, (unsigned long long)K, 64u, 128u);
        }
#else
#pragma unroll
        for (int c = 0; c < 2; ++c) {
            const int chunk = c * 256 + tid;          // 512 chunks of 16
            const int r = chunk >> 2, c0 = (chunk & 3) * 16;
            const U128* ga = (const U128*)(A + (size_t)(bm + r) * K + k0 + c0);
            *(U128*)(As + r * 64 + c0)     = ga[0];
            *(U128*)(As + r * 64 + c0 + 8) = ga[1];
        }
#endif
        // ---- stage B tile 64x128 transposed into [n][k] ----
#pragma unroll
        for (int c = 0; c < 2; ++c) {
            const int chunk = c * 256 + tid;          // 512 chunks of 16
            const int kr = chunk >> 3, n0 = (chunk & 7) * 16;
            const U128* gb = (const U128*)(B + (size_t)(k0 + kr) * N + bn + n0);
            Pack p; p.q[0] = gb[0]; p.q[1] = gb[1];
#pragma unroll
            for (int j = 0; j < 16; ++j) Bs[(n0 + j) * 64 + kr] = p.s[j];
        }
#if USE_TDM
        if (tid < 32) __builtin_amdgcn_s_wait_tensorcnt(0);
#endif
        __syncthreads();

        if (k0 + 64 < K) {   // CDNA5 global_prefetch_b8 for next B tile
            __builtin_prefetch(B + (size_t)(k0 + 64 + (tid >> 3)) * N + bn + (tid & 7) * 16, 0, 0);
        }

        // ---- 16 WMMAs per wave per stage ----
#pragma unroll
        for (int kk = 0; kk < 64; kk += 32) {
            v16bf bf[2];
#pragma unroll
            for (int nt = 0; nt < 2; ++nt)
                bf[nt] = frag_b(Bs, wn * 32 + nt * 16 + (lane & 15), 64, kk);
#pragma unroll
            for (int mt = 0; mt < 4; ++mt) {
                v16bf af = frag_a(As, wm * 64 + mt * 16 + (lane & 15), 64, kk);
#pragma unroll
                for (int nt = 0; nt < 2; ++nt)
                    acc[mt][nt] = wmma_bf16(af, bf[nt], acc[mt][nt]);
            }
        }
        __syncthreads();
    }

    // ---- epilogue (C layout: vgpr r -> M=r / r+8, lane&15 -> N) ----
#pragma unroll
    for (int mt = 0; mt < 4; ++mt)
#pragma unroll
        for (int nt = 0; nt < 2; ++nt)
#pragma unroll
            for (int r = 0; r < 8; ++r) {
                const int m   = (lane < 16) ? r : (r + 8);
                const int row = bm + wm * 64 + mt * 16 + m;
                const int col = bn + wn * 32 + nt * 16 + (lane & 15);
                const size_t idx = (size_t)row * N + col;
                float v = acc[mt][nt][r] + bias[col];
                if constexpr (EPI == 1) {
                    ((float*)Cout)[idx] = v + resid[idx];
                } else {
                    ((unsigned short*)Cout)[idx] = f2bf(v);
                }
            }
}

// ---------------------------------------------------------------------------
// Block attention with per-(i,j)-block softmax (matches reference semantics).
// One workgroup = (b, h, query-block i, 32-row quarter). 8 waves, each wave
// owns 16 output columns across all 32 rows (2x1 WMMA accs).
// qkv layout: [row = b*2048+s][which*1024 + h*128 + d], bf16, stride 3072.
// ---------------------------------------------------------------------------
__global__ __launch_bounds__(256) void attn_kernel(
    const unsigned short* __restrict__ qkv, unsigned short* __restrict__ attnout)
{
    __shared__ unsigned short Qs[32 * 128];     //  8 KB
    __shared__ unsigned short KVs[128 * 128];   // 32 KB (K tile, then V^T tile)
    __shared__ float Ss[32 * 128];              // 16 KB scores; low half reused as P bf16
    unsigned short* Ps = (unsigned short*)Ss;

    const int tid  = threadIdx.x;
    const int lane = tid & 31;
    const int w    = tid >> 5;                  // wave id: output cols w*16..+15
    const int bid  = blockIdx.x;
    const int q4   = bid & 3;                   // quarter (32 query rows)
    const int ib   = (bid >> 2) & 15;           // query block
    const int hh   = (bid >> 6) & 7;            // head
    const int bb   = bid >> 9;                  // batch

    const size_t rs = 3072;
    const int    hc = hh * 128;

    // ---- load Q quarter, scaled by 1/sqrt(HD) ----
    {
        const int r = tid >> 3, c0 = (tid & 7) * 16;
        const size_t grow = (size_t)(bb * 2048 + ib * 128 + q4 * 32 + r);
        const unsigned short* src = qkv + grow * rs + hc + c0;
#pragma unroll
        for (int j = 0; j < 16; ++j)
            Qs[r * 128 + c0 + j] = f2bf(bf2f(src[j]) * 0.08838834764831845f);
    }
    __syncthreads();

    const v8f vz = {0.f, 0.f, 0.f, 0.f, 0.f, 0.f, 0.f, 0.f};
    v8f o[2] = {vz, vz};                        // persistent output accumulators

    for (int jb = 0; jb <= ib; ++jb) {
        // ---- stage K_j tile, natural [s][d] layout ----
#if USE_TDM
        if (tid < 32) {
            tdm_load_2d((unsigned)(unsigned long long)(void*)KVs,
                        qkv + (size_t)(bb * 2048 + jb * 128) * rs + 1024 + hc,
                        128u, 128u, 3072ull, 128u, 128u);
            __builtin_amdgcn_s_wait_tensorcnt(0);
        }
#else
#pragma unroll
        for (int it = 0; it < 4; ++it) {
            const int cidx = it * 256 + tid;
            const int sr = cidx >> 3, c0 = (cidx & 7) * 16;
            const U128* g = (const U128*)(qkv +
                (size_t)(bb * 2048 + jb * 128 + sr) * rs + 1024 + hc + c0);
            *(U128*)(KVs + sr * 128 + c0)     = g[0];
            *(U128*)(KVs + sr * 128 + c0 + 8) = g[1];
        }
#endif
        __syncthreads();

        // ---- S = Q K^T : B-frag column = key s, K-dim = d (contiguous) ----
        v8f s[2] = {vz, vz};
#pragma unroll
        for (int k0 = 0; k0 < 128; k0 += 32) {
            v16bf bf = frag_b(KVs, w * 16 + (lane & 15), 128, k0);
#pragma unroll
            for (int mt = 0; mt < 2; ++mt) {
                v16bf af = frag_a(Qs, mt * 16 + (lane & 15), 128, k0);
                s[mt] = wmma_bf16(af, bf, s[mt]);
            }
        }
        // causal mask on diagonal block only
        if (jb == ib) {
            const int kloc = w * 16 + (lane & 15);
#pragma unroll
            for (int mt = 0; mt < 2; ++mt)
#pragma unroll
                for (int r = 0; r < 8; ++r) {
                    const int m = (lane < 16) ? r : (r + 8);
                    if (kloc > q4 * 32 + mt * 16 + m) s[mt][r] = -1e10f;
                }
        }
        // scatter scores to LDS
#pragma unroll
        for (int mt = 0; mt < 2; ++mt)
#pragma unroll
            for (int r = 0; r < 8; ++r) {
                const int m = (lane < 16) ? r : (r + 8);
                Ss[(mt * 16 + m) * 128 + w * 16 + (lane & 15)] = s[mt][r];
            }
        __syncthreads();

        // ---- per-row softmax over this 128-key block (8 lanes per row) ----
        {
            const int sr = tid >> 3, c0 = (tid & 7) * 16;
            float vbuf[16];
#pragma unroll
            for (int j = 0; j < 16; ++j) vbuf[j] = Ss[sr * 128 + c0 + j];
            float mx = vbuf[0];
#pragma unroll
            for (int j = 1; j < 16; ++j) mx = fmaxf(mx, vbuf[j]);
#pragma unroll
            for (int msk = 1; msk <= 4; msk <<= 1)
                mx = fmaxf(mx, __shfl_xor(mx, msk, 32));
            float e[16], sum = 0.f;
#pragma unroll
            for (int j = 0; j < 16; ++j) { e[j] = __expf(vbuf[j] - mx); sum += e[j]; }
#pragma unroll
            for (int msk = 1; msk <= 4; msk <<= 1)
                sum += __shfl_xor(sum, msk, 32);
            const float inv = 1.0f / sum;
            __syncthreads();                    // everyone done reading Ss
#pragma unroll
            for (int j = 0; j < 16; ++j)
                Ps[sr * 128 + c0 + j] = f2bf(e[j] * inv);
        }
        __syncthreads();

        // ---- stage V_j transposed: KVs[d][s] = V[s][d] ----
#pragma unroll
        for (int it = 0; it < 4; ++it) {
            const int cidx = it * 256 + tid;
            const int vr = cidx >> 3, c0v = (cidx & 7) * 16;
            const U128* g = (const U128*)(qkv +
                (size_t)(bb * 2048 + jb * 128 + vr) * rs + 2048 + hc + c0v);
            Pack p; p.q[0] = g[0]; p.q[1] = g[1];
#pragma unroll
            for (int j = 0; j < 16; ++j) KVs[(c0v + j) * 128 + vr] = p.s[j];
        }
        __syncthreads();

        // ---- O += P V : B-frag column = d, K-dim = s (contiguous in V^T) ----
#pragma unroll
        for (int k0 = 0; k0 < 128; k0 += 32) {
            v16bf bf = frag_b(KVs, w * 16 + (lane & 15), 128, k0);
#pragma unroll
            for (int mt = 0; mt < 2; ++mt) {
                v16bf af = frag_a(Ps, mt * 16 + (lane & 15), 128, k0);
                o[mt] = wmma_bf16(af, bf, o[mt]);
            }
        }
        __syncthreads();                        // before next jb clobbers KVs/Ps
    }

    // ---- store O as bf16 into [row][h*128 + d] context layout ----
#pragma unroll
    for (int mt = 0; mt < 2; ++mt)
#pragma unroll
        for (int r = 0; r < 8; ++r) {
            const int m    = (lane < 16) ? r : (r + 8);
            const int grow = bb * 2048 + ib * 128 + q4 * 32 + mt * 16 + m;
            const int gcol = hc + w * 16 + (lane & 15);
            attnout[(size_t)grow * 1024 + gcol] = f2bf(o[mt][r]);
        }
}

// ---------------------------------------------------------------------------
// LayerNorm over D=1024, one wave per row, bf16 output.
// ---------------------------------------------------------------------------
__global__ __launch_bounds__(256) void ln_kernel(
    const float* __restrict__ x, const float* __restrict__ gamma,
    const float* __restrict__ beta, unsigned short* __restrict__ out)
{
    const int lane = threadIdx.x & 31;
    const int row  = blockIdx.x * 8 + (threadIdx.x >> 5);
    const size_t base = (size_t)row * 1024;
    float v[32], sum = 0.f;
#pragma unroll
    for (int j = 0; j < 32; ++j) { v[j] = x[base + j * 32 + lane]; sum += v[j]; }
#pragma unroll
    for (int m = 16; m >= 1; m >>= 1) sum += __shfl_xor(sum, m, 32);
    const float mean = sum * (1.0f / 1024.0f);
    float var = 0.f;
#pragma unroll
    for (int j = 0; j < 32; ++j) { const float d = v[j] - mean; var += d * d; }
#pragma unroll
    for (int m = 16; m >= 1; m >>= 1) var += __shfl_xor(var, m, 32);
    const float rstd = rsqrtf(var * (1.0f / 1024.0f) + 1e-5f);
#pragma unroll
    for (int j = 0; j < 32; ++j) {
        const int c = j * 32 + lane;
        out[base + c] = f2bf((v[j] - mean) * rstd * gamma[c] + beta[c]);
    }
}

// ---------------------------------------------------------------------------
// f32 -> bf16 weight conversion
// ---------------------------------------------------------------------------
__global__ __launch_bounds__(256) void cvt_kernel(
    const float* __restrict__ in, unsigned short* __restrict__ out, int n)
{
    const int i = blockIdx.x * 256 + threadIdx.x;
    if (i < n) out[i] = f2bf(in[i]);
}

// ---------------------------------------------------------------------------
// SwiGLU: mid[m, c] = swish(up[m, c]) * up[m, 4096 + c]
// ---------------------------------------------------------------------------
__global__ __launch_bounds__(256) void swiglu_kernel(
    const unsigned short* __restrict__ up, unsigned short* __restrict__ mid)
{
    const int i = blockIdx.x * 256 + threadIdx.x;     // 4096*4096 total
    const int m = i >> 12, c = i & 4095;
    const float g  = bf2f(up[(size_t)m * 8192 + c]);
    const float vv = bf2f(up[(size_t)m * 8192 + 4096 + c]);
    const float sw = g / (1.0f + __expf(-g));
    mid[i] = f2bf(sw * vv);
}

// ---------------------------------------------------------------------------
// Host launcher
// ---------------------------------------------------------------------------
extern "C" void kernel_launch(void* const* d_in, const int* in_sizes, int n_in,
                              void* d_out, int out_size, void* d_ws, size_t ws_size,
                              hipStream_t stream) {
    (void)in_sizes; (void)n_in; (void)out_size; (void)ws_size;
    const float* x     = (const float*)d_in[0];
    const float* ln1g  = (const float*)d_in[1];
    const float* ln1b  = (const float*)d_in[2];
    const float* qkvw  = (const float*)d_in[3];
    const float* qkvb  = (const float*)d_in[4];
    const float* outw  = (const float*)d_in[5];
    const float* outb  = (const float*)d_in[6];
    const float* ln2g  = (const float*)d_in[7];
    const float* ln2b  = (const float*)d_in[8];
    const float* upw   = (const float*)d_in[9];
    const float* upb   = (const float*)d_in[10];
    const float* downw = (const float*)d_in[11];
    const float* downb = (const float*)d_in[12];
    float* outp = (float*)d_out;

    char* ws = (char*)d_ws;
    size_t off = 0;
    auto alloc = [&](size_t bytes) -> void* {
        void* p = ws + off;
        off += (bytes + 255) & ~(size_t)255;
        return p;
    };
    unsigned short* qkvw_bf  = (unsigned short*)alloc(1024ull * 3072 * 2);
    unsigned short* outw_bf  = (unsigned short*)alloc(1024ull * 1024 * 2);
    unsigned short* upw_bf   = (unsigned short*)alloc(1024ull * 8192 * 2);
    unsigned short* downw_bf = (unsigned short*)alloc(4096ull * 1024 * 2);
    unsigned short* h1       = (unsigned short*)alloc(4096ull * 1024 * 2);
    unsigned short* qkvbuf   = (unsigned short*)alloc(4096ull * 3072 * 2);
    unsigned short* attnbuf  = (unsigned short*)alloc(4096ull * 1024 * 2);
    float*          x1       = (float*)         alloc(4096ull * 1024 * 4);
    unsigned short* h2       = (unsigned short*)alloc(4096ull * 1024 * 2);
    unsigned short* upbuf    = (unsigned short*)alloc(4096ull * 8192 * 2);
    unsigned short* midbuf   = (unsigned short*)alloc(4096ull * 4096 * 2);

    // weights -> bf16
    cvt_kernel<<<(1024 * 3072 + 255) / 256, 256, 0, stream>>>(qkvw,  qkvw_bf,  1024 * 3072);
    cvt_kernel<<<(1024 * 1024 + 255) / 256, 256, 0, stream>>>(outw,  outw_bf,  1024 * 1024);
    cvt_kernel<<<(1024 * 8192 + 255) / 256, 256, 0, stream>>>(upw,   upw_bf,   1024 * 8192);
    cvt_kernel<<<(4096 * 1024 + 255) / 256, 256, 0, stream>>>(downw, downw_bf, 4096 * 1024);

    // h1 = LN1(x)
    ln_kernel<<<512, 256, 0, stream>>>(x, ln1g, ln1b, h1);
    // qkv = h1 @ qkv_w + qkv_b   (bf16 out)
    gemm_bf16_kernel<0><<<dim3(24, 32), 256, 0, stream>>>(
        h1, qkvw_bf, qkvb, nullptr, qkvbuf, 4096, 3072, 1024);
    // block attention (per-block softmax, diag causal mask, j<=i)
    attn_kernel<<<1024, 256, 0, stream>>>(qkvbuf, attnbuf);
    // x1 = x + attn @ out_w + out_b   (f32 out, fused residual)
    gemm_bf16_kernel<1><<<dim3(8, 32), 256, 0, stream>>>(
        attnbuf, outw_bf, outb, x, x1, 4096, 1024, 1024);
    // h2 = LN2(x1)
    ln_kernel<<<512, 256, 0, stream>>>(x1, ln2g, ln2b, h2);
    // up = h2 @ up_w + up_b   (bf16 out)
    gemm_bf16_kernel<0><<<dim3(64, 32), 256, 0, stream>>>(
        h2, upw_bf, upb, nullptr, upbuf, 4096, 8192, 1024);
    // mid = swish(gate) * val
    swiglu_kernel<<<(4096 * 4096) / 256, 256, 0, stream>>>(upbuf, midbuf);
    // out = x1 + mid @ down_w + down_b   (f32 out, fused residual)
    gemm_bf16_kernel<1><<<dim3(8, 32), 256, 0, stream>>>(
        midbuf, downw_bf, downb, x1, outp, 4096, 1024, 4096);
}
